// ConvNet_2362232013296
// MI455X (gfx1250) — matrix operand order repaired and checked by reading, not compiled, not run on hardware
//
#include <hip/hip_runtime.h>
#include <hip/hip_bf16.h>

// ---------------------------------------------------------------------------
// MI455X (gfx1250) spiking-CNN recurrence — WMMA implicit-GEMM, v4.
//
// Roofline: ~109 GFLOP over T=32 steps; unique HBM traffic is the input x
// (~100 MB, one pass). LIF state + weights are L2-resident (192 MB L2), so
// the job is keeping v_wmma_f32_16x16x32_f16 fed from L2.
// v4 vs v3 (asm-driven): padded-K gather masks become unconditional loads +
// v_cndmask selects (delta table stores 0 for pad-k, so the address is always
// valid) instead of per-element s_and_saveexec/s_cbranch blocks. FC tail
// fragment fill likewise select-based.
// FP16 A/B with F32 accumulate: spikes are exactly {0,1}, weights ~N(0,1/fan).
// ---------------------------------------------------------------------------

typedef __attribute__((ext_vector_type(16))) _Float16 v16h;
typedef __attribute__((ext_vector_type(8)))  _Float16 v8h;
typedef __attribute__((ext_vector_type(8)))  float    v8f;

#define TB   32
#define BB   64
#define P1   688   // padded 26*26=676 plane (multiple of 16)
// LIF constants: DT*TAU_MEM_INV = 0.1, 1 - DT*TAU_SYN_INV = 0.8
#define A_MEM 0.1f
#define A_SYN 0.8f

// ISA 16-bit A-matrix 16x32 layout (CDNA5 ISA):
// lane group h = lane>>4; j<8 -> K = h*8+j ; j>=8 -> K = 16 + h*8 + (j-8)
__device__ __forceinline__ int amap_k(int h, int j) {
  return (j < 8) ? (h * 8 + j) : (16 + h * 8 + (j - 8));
}
// Contiguous k order (this code's choice, not the ISA table): lane half h holds K=16h..16h+15; fc_lif uses it for A and B.
__device__ __forceinline__ int bmap_k(int h, int j) { return h * 16 + j; }

// Scalar fused LIF update + spike store.
__device__ __forceinline__ void lif_store(float inp, float* __restrict__ v,
                                          float* __restrict__ i,
                                          _Float16* __restrict__ z, long idx) {
  float vv = v[idx], ii = i[idx];
  float vdec = vv + A_MEM * (ii - vv);
  float spike = vdec > 1.0f ? 1.0f : 0.0f;
  v[idx] = (1.0f - spike) * vdec;
  i[idx] = A_SYN * ii + inp;
  z[idx] = (_Float16)spike;
}

// Vectorized 8-wide LIF; base must be a multiple of 8 elements.
__device__ __forceinline__ void lif8(const v8f& acc, float scale, float bias,
                                     float* __restrict__ v,
                                     float* __restrict__ i,
                                     _Float16* __restrict__ z, long base) {
  float4 va = *(const float4*)(v + base);
  float4 vb = *(const float4*)(v + base + 4);
  float4 ia = *(const float4*)(i + base);
  float4 ib = *(const float4*)(i + base + 4);
  float vv[8] = {va.x, va.y, va.z, va.w, vb.x, vb.y, vb.z, vb.w};
  float ii[8] = {ia.x, ia.y, ia.z, ia.w, ib.x, ib.y, ib.z, ib.w};
  v8h zz;
#pragma unroll
  for (int r = 0; r < 8; ++r) {
    float vdec = vv[r] + A_MEM * (ii[r] - vv[r]);
    float spike = vdec > 1.0f ? 1.0f : 0.0f;
    vv[r] = (1.0f - spike) * vdec;
    ii[r] = A_SYN * ii[r] + scale * (acc[r] + bias);
    zz[r] = (_Float16)spike;
  }
  *(float4*)(v + base)     = make_float4(vv[0], vv[1], vv[2], vv[3]);
  *(float4*)(v + base + 4) = make_float4(vv[4], vv[5], vv[6], vv[7]);
  *(float4*)(i + base)     = make_float4(ii[0], ii[1], ii[2], ii[3]);
  *(float4*)(i + base + 4) = make_float4(ii[4], ii[5], ii[6], ii[7]);
  *(v8h*)(z + base) = zz;
}

// --------------------------- state zero-init -------------------------------
__global__ void zero_f32(float* __restrict__ p, long n) {
  long idx = (long)blockIdx.x * blockDim.x + threadIdx.x;
  if (idx < n) p[idx] = 0.0f;
}

// ------------------------------ conv1 + LIF0 -------------------------------
// x_t: [B,3,64,64] f32. Per image M=3600, N=20(pad 32), K=75(pad 96).
// Wave = 16x32 tile: 2 WMMA x 3 k-steps.
__global__ __launch_bounds__(128) void conv1_lif(
    const float* __restrict__ x, const float* __restrict__ w1,
    const float* __restrict__ b1, float* __restrict__ v0,
    float* __restrict__ i0, _Float16* __restrict__ z0) {
  __shared__ __align__(32) _Float16 Bsh[3 * 2 * 32 * 16];
  __shared__ __align__(16) int dlt[96];
  const int tid = threadIdx.x, lane = tid & 31, wid = tid >> 5;
  const int b = blockIdx.x / 57, tg = blockIdx.x % 57;
  const int tile = tg * 4 + wid;  // 0..227, valid < 225

#pragma unroll 1
  for (int k = tid; k < 96; k += 128) {
    int c = k / 25, rem = k % 25;
    dlt[k] = (k < 75) ? ((c * 64 + rem / 5) * 64 + rem % 5) : 0;
  }
#pragma unroll 1
  for (int idx = tid; idx < 3 * 2 * 32 * 16; idx += 128) {
    int j = idx & 15, l = (idx >> 4) & 31, nh = (idx >> 9) & 1, ks = idx >> 10;
    int k = ks * 32 + amap_k(l >> 4, j);
    int n = nh * 16 + (l & 15);
    Bsh[idx] = (_Float16)((n < 20 && k < 75) ? w1[n * 75 + k] : 0.0f);
  }
  __syncthreads();

  const int hi = lane >> 4, khi = hi * 8;
  int p = tile * 16 + (lane & 15);
  int pc = p < 3600 ? p : 3599;
  const int ibase = (pc / 60) * 64 + (pc % 60);   // lane's 32-bit offset
  const float* xb = x + (long)b * (3 * 64 * 64);  // block-uniform base (SGPR)

  v8f acc0 = {}, acc1 = {};
#pragma unroll
  for (int ks = 0; ks < 3; ++ks) {
    int dk[16];
    {
      const int4* dp = (const int4*)&dlt[ks * 32 + khi];
      const int4* dq = (const int4*)&dlt[ks * 32 + khi + 16];
      int4 d0 = dp[0], d1 = dp[1], d2 = dq[0], d3 = dq[1];
      dk[0] = d0.x; dk[1] = d0.y; dk[2] = d0.z; dk[3] = d0.w;
      dk[4] = d1.x; dk[5] = d1.y; dk[6] = d1.z; dk[7] = d1.w;
      dk[8] = d2.x; dk[9] = d2.y; dk[10] = d2.z; dk[11] = d2.w;
      dk[12] = d3.x; dk[13] = d3.y; dk[14] = d3.z; dk[15] = d3.w;
    }
    v16h a;
#pragma unroll
    for (int j = 0; j < 16; ++j) {
      float v;
      if (ks < 2) {
        v = xb[ibase + dk[j]];   // k <= 63 < 75: never padded
      } else if (j < 8) {
        // k = 64+khi+j; pad-k has dlt==0 so the load is always safe:
        // unconditional load + v_cndmask (no exec-branch around the load).
        float raw = xb[ibase + dk[j]];
        v = (khi + j < 11) ? raw : 0.0f;
      } else {
        v = 0.0f;                // k >= 80: always padded
      }
      a[j] = (_Float16)v;
    }
    v16h bf0 = *(const v16h*)&Bsh[((ks * 2 + 0) * 32 + lane) * 16];
    v16h bf1 = *(const v16h*)&Bsh[((ks * 2 + 1) * 32 + lane) * 16];
    acc0 = __builtin_amdgcn_wmma_f32_16x16x32_f16(false, a, false, bf0,
                                                  (short)0, acc0, false, false);
    acc1 = __builtin_amdgcn_wmma_f32_16x16x32_f16(false, a, false, bf1,
                                                  (short)0, acc1, false, false);
  }

  const int n0 = lane & 15, mbase = hi * 8;
  const int pbase = tile * 16 + mbase;       // multiple of 8; 3600 % 8 == 0
  if (pbase < 3600) {
    lif8(acc0, 1.0f, b1[n0], v0, i0, z0, (long)(b * 20 + n0) * 3600 + pbase);
    if (n0 + 16 < 20)
      lif8(acc1, 1.0f, b1[n0 + 16], v0, i0, z0,
           (long)(b * 20 + n0 + 16) * 3600 + pbase);
  }
}

// ------------------------------- 2x2 maxpool -------------------------------
// Spikes are {0,1}: max == OR. One (b,c) plane per blockIdx.y; PS = input
// plane stride in elements (688 for the padded conv2 spike planes).
__global__ void pool2(const _Float16* __restrict__ z, _Float16* __restrict__ s,
                      int H, int PS) {
  int Ho = H >> 1;
  int sp = blockIdx.x * blockDim.x + threadIdx.x;
  if (sp >= Ho * Ho) return;
  long plane = blockIdx.y;
  int py = sp / Ho, px = sp % Ho;
  const _Float16* base = z + plane * PS + (2 * py) * H + 2 * px;
  float m = fmaxf(fmaxf((float)base[0], (float)base[1]),
                  fmaxf((float)base[H], (float)base[H + 1]));
  s[plane * Ho * Ho + sp] = (_Float16)m;
}

// ------------------------------ conv2 + LIF1 -------------------------------
// s0: [B,20,30,30] f16. Per image M=676(pad 688), N=50(pad 64), K=500(pad
// 512). Wave = 16x64 tile (4 accumulators), 16 k-steps, weights restaged per
// k-step in LDS (w2 = 100 KB, L2-resident). State planes padded to 688.
__global__ __launch_bounds__(128) void conv2_lif(
    const _Float16* __restrict__ s0, const float* __restrict__ w2,
    const float* __restrict__ b2, float* __restrict__ v1,
    float* __restrict__ i1, _Float16* __restrict__ z1) {
  __shared__ __align__(32) _Float16 Bsh[4 * 32 * 16];
  __shared__ __align__(16) int dlt[512];
  const int tid = threadIdx.x, lane = tid & 31, wid = tid >> 5;
  const int b = blockIdx.x / 11, tg = blockIdx.x % 11;
  const int tile = tg * 4 + wid;  // 0..43
  const int hi = lane >> 4, khi = hi * 8;
  int p = tile * 16 + (lane & 15);
  int pc = p < 676 ? p : 675;
  const int ibase = (pc / 26) * 30 + (pc % 26);        // 32-bit lane offset
  const _Float16* sb = s0 + (long)b * (20 * 30 * 30);  // uniform base (SGPR)

#pragma unroll 1
  for (int k = tid; k < 512; k += 128) {
    int c = k / 25, rem = k % 25;
    dlt[k] = (k < 500) ? ((c * 30 + rem / 5) * 30 + rem % 5) : 0;
  }

  v8f acc[4] = {};
#pragma unroll 1
  for (int ks = 0; ks < 16; ++ks) {
#pragma unroll 1
    for (int idx = tid; idx < 4 * 32 * 16; idx += 128) {
      int j = idx & 15, l = (idx >> 4) & 31, nh = idx >> 9;
      int k = ks * 32 + amap_k(l >> 4, j);
      int n = nh * 16 + (l & 15);
      Bsh[idx] = (_Float16)((n < 50 && k < 500) ? w2[n * 500 + k] : 0.0f);
    }
    __syncthreads();
    int dk[16];
    {
      const int4* dp = (const int4*)&dlt[ks * 32 + khi];
      const int4* dq = (const int4*)&dlt[ks * 32 + khi + 16];
      int4 d0 = dp[0], d1 = dp[1], d2 = dq[0], d3 = dq[1];
      dk[0] = d0.x; dk[1] = d0.y; dk[2] = d0.z; dk[3] = d0.w;
      dk[4] = d1.x; dk[5] = d1.y; dk[6] = d1.z; dk[7] = d1.w;
      dk[8] = d2.x; dk[9] = d2.y; dk[10] = d2.z; dk[11] = d2.w;
      dk[12] = d3.x; dk[13] = d3.y; dk[14] = d3.z; dk[15] = d3.w;
    }
    v16h a;
    if (ks < 15) {  // k <= 479 < 500: never padded
#pragma unroll
      for (int j = 0; j < 16; ++j) a[j] = sb[ibase + dk[j]];
    } else {
      // k = 480 + amap; pad-k has dlt==0 so loads are always safe:
      // unconditional load + select instead of exec-branch per element.
#pragma unroll
      for (int j = 0; j < 16; ++j) {
        _Float16 raw = sb[ibase + dk[j]];
        bool valid = (j < 8) || (khi + j < 12);
        a[j] = valid ? raw : (_Float16)0.0f;
      }
    }
#pragma unroll
    for (int nh = 0; nh < 4; ++nh) {
      v16h bf = *(const v16h*)&Bsh[(nh * 32 + lane) * 16];
      acc[nh] = __builtin_amdgcn_wmma_f32_16x16x32_f16(
          false, a, false, bf, (short)0, acc[nh], false, false);
    }
    __syncthreads();
  }

  const int mbase = hi * 8;
  const int pbase = tile * 16 + mbase;  // multiple of 8
  if (pbase < 676) {                    // pad cells [676,680) written, unread
#pragma unroll
    for (int nh = 0; nh < 4; ++nh) {
      int n = nh * 16 + (lane & 15);
      if (n < 50)  // reference scales conv2 output (incl. bias) by 10
        lif8(acc[nh], 10.0f, b2[n], v1, i1, z1,
             (long)(b * 50 + n) * P1 + pbase);
    }
  }
}

// ------------------------------- FC + LIF2 ---------------------------------
// s1: [64, 8450] f16. GEMM M=64, N=500(pad 512), K=8450 = 264 clean k-steps +
// 2-element tail. Block = 4 waves sharing one A tile in LDS; each wave owns a
// 16-col N tile. ReLU after LIF is the identity on {0,1} spikes.
__global__ __launch_bounds__(128) void fc_lif(
    const _Float16* __restrict__ s1, const float* __restrict__ wf,
    const float* __restrict__ bf, float* __restrict__ v2,
    float* __restrict__ i2, _Float16* __restrict__ z2) {
  __shared__ __align__(32) _Float16 Ash[32 * 16];
  const int tid = threadIdx.x, lane = tid & 31, wid = tid >> 5;
  const int mt = blockIdx.x >> 3;              // 4 batch tiles of 16
  const int ng = blockIdx.x & 7;               // 8 groups of 4 N-tiles
  const int ncol = (ng * 4 + wid) * 16 + (lane & 15);
  const int ncol_eff = ncol < 500 ? ncol : 499;  // clamp for safe vector loads
  const int hi = lane >> 4;
  const _Float16* srow = s1 + (long)(mt * 16) * 8450;
  const float* wrow = wf + (long)ncol_eff * 8450;  // 8B-aligned rows

  v8f acc = {};
#pragma unroll 1
  for (int ks = 0; ks < 264; ++ks) {  // k <= 263*32+31 = 8447 < 8450
#pragma unroll 1
    for (int idx = tid; idx < 512; idx += 128) {
      int j = idx & 15, l = idx >> 4;
      int k = ks * 32 + bmap_k(l >> 4, j);
      Ash[idx] = srow[(l & 15) * 8450 + k];
    }
    __syncthreads();
    v16h a = *(const v16h*)&Ash[lane * 16];
    const float2* wp = (const float2*)(wrow + ks * 32 + hi * 16);
    v16h bb;
#pragma unroll
    for (int j2 = 0; j2 < 8; ++j2) {
      float2 w2v = wp[j2];
      bb[2 * j2] = (_Float16)w2v.x;
      bb[2 * j2 + 1] = (_Float16)w2v.y;
    }
    acc = __builtin_amdgcn_wmma_f32_16x16x32_f16(false, a, false, bb, (short)0,
                                                 acc, false, false);
    __syncthreads();
  }
  {  // tail k-step: only k = 8448, 8449 are live
#pragma unroll 1
    for (int idx = tid; idx < 512; idx += 128) {
      int j = idx & 15, l = idx >> 4;
      int k = 8448 + bmap_k(l >> 4, j);
      int kc = k < 8450 ? k : 8448;  // safe address; select below
      _Float16 raw = srow[(l & 15) * 8450 + kc];
      Ash[idx] = (k < 8450) ? raw : (_Float16)0.0f;
    }
    __syncthreads();
    v16h a = *(const v16h*)&Ash[lane * 16];
    // Only lanes 0-15 (hi==0) carry live K in this step: select, not branch.
    float r0 = wrow[8448], r1 = wrow[8449];
    v16h bb = {};
    bb[0] = (hi == 0) ? (_Float16)r0 : (_Float16)0.0f;
    bb[1] = (hi == 0) ? (_Float16)r1 : (_Float16)0.0f;
    acc = __builtin_amdgcn_wmma_f32_16x16x32_f16(false, a, false, bb, (short)0,
                                                 acc, false, false);
  }
  if (ncol < 500) {
    float bias = bf[ncol];
    int mbase = hi * 8;
#pragma unroll
    for (int r = 0; r < 8; ++r) {
      int batch = mt * 16 + mbase + r;
      lif_store(acc[r] + bias, v2, i2, z2, (long)batch * 500 + ncol);
    }
  }
}

// ------------------------- output leaky integrator -------------------------
// 64x10 from 64x500 spikes: 320K MACs — VALU is fine, WMMA would be waste.
__global__ __launch_bounds__(640) void out_li(
    const _Float16* __restrict__ z2, const float* __restrict__ wo,
    const float* __restrict__ bo, float* __restrict__ vo,
    float* __restrict__ io, float* __restrict__ out_t) {
  int tid = threadIdx.x;
  int b = tid / 10, o = tid % 10;
  const _Float16* zb = z2 + b * 500;
  const float* w = wo + o * 500;
  float dot = 0.0f;
#pragma unroll 1
  for (int j = 0; j < 500; ++j) dot += (float)zb[j] * w[j];
  float vv = vo[tid], ii = io[tid];
  float vn = vv + A_MEM * (ii - vv);  // voltage update uses OLD current
  io[tid] = A_SYN * ii + dot + bo[o];
  vo[tid] = vn;
  out_t[tid] = vn;
}

// ---------------------------------------------------------------------------
extern "C" void kernel_launch(void* const* d_in, const int* in_sizes, int n_in,
                              void* d_out, int out_size, void* d_ws,
                              size_t ws_size, hipStream_t stream) {
  const float* x  = (const float*)d_in[0];
  const float* w1 = (const float*)d_in[1];
  const float* b1 = (const float*)d_in[2];
  const float* w2 = (const float*)d_in[3];
  const float* b2 = (const float*)d_in[4];
  const float* wf = (const float*)d_in[5];
  const float* bf = (const float*)d_in[6];
  const float* wo = (const float*)d_in[7];
  const float* bo = (const float*)d_in[8];
  float* out = (float*)d_out;

  const long N0  = (long)BB * 20 * 3600;  // 4,608,000
  const long N1P = (long)BB * 50 * P1;    // 2,201,600 (padded planes)
  const long N2  = (long)BB * 500;        // 32,000
  const long NO  = (long)BB * 10;         // 640
  float* f = (float*)d_ws;
  float* v0 = f;            f += N0;
  float* i0 = f;            f += N0;
  float* v1 = f;            f += N1P;
  float* i1 = f;            f += N1P;
  float* v2 = f;            f += N2;
  float* i2 = f;            f += N2;
  float* vo = f;            f += NO;
  float* io = f;            f += NO;
  const long STATE = 2 * (N0 + N1P + N2 + NO);
  _Float16* h = (_Float16*)f;
  _Float16* z0 = h;         h += N0;                       // [B,20,60,60]
  _Float16* s0 = h;         h += (long)BB * 20 * 30 * 30;  // pooled spikes
  _Float16* z1 = h;         h += N1P;                      // [B,50,688pad]
  _Float16* s1 = h;         h += (long)BB * 8450;          // flattened pooled
  _Float16* z2 = h;         h += N2;                       // [B,500]

  // Deterministic: reset recurrent state every call.
  zero_f32<<<dim3((unsigned)((STATE + 255) / 256)), dim3(256), 0, stream>>>(
      (float*)d_ws, STATE);

  for (int t = 0; t < TB; ++t) {
    const float* xt = x + (long)t * BB * 3 * 64 * 64;
    conv1_lif<<<dim3(BB * 57), dim3(128), 0, stream>>>(xt, w1, b1, v0, i0, z0);
    pool2<<<dim3(4, BB * 20), dim3(256), 0, stream>>>(z0, s0, 60, 3600);
    conv2_lif<<<dim3(BB * 11), dim3(128), 0, stream>>>(s0, w2, b2, v1, i1, z1);
    pool2<<<dim3(1, BB * 50), dim3(256), 0, stream>>>(z1, s1, 26, P1);
    fc_lif<<<dim3(32), dim3(128), 0, stream>>>(s1, wf, bf, v2, i2, z2);
    out_li<<<dim3(1), dim3(640), 0, stream>>>(z2, wo, bo, vo, io,
                                              out + (long)t * BB * 10);
  }
}
